// BatchProgramCC_13804024889631
// MI455X (gfx1250) — matrix operand order repaired and checked by reading, not compiled, not run on hardware
//
#include <hip/hip_runtime.h>
#include <math.h>

typedef __attribute__((ext_vector_type(16))) __bf16 v16bf;
typedef __attribute__((ext_vector_type(8)))  float  v8f;

#define T_TREES 2048
#define P_NODES 256
#define E_DIM   128
#define C_DIM   128
#define H_DIM   128
#define G3      384   // 3*H

__device__ __forceinline__ float sigmoidf_(float x) { return 1.0f / (1.0f + expf(-x)); }

__device__ __forceinline__ void pack8_bf16(v16bf& v, int base, float4 a, float4 b) {
    v[base + 0] = (__bf16)a.x; v[base + 1] = (__bf16)a.y;
    v[base + 2] = (__bf16)a.z; v[base + 3] = (__bf16)a.w;
    v[base + 4] = (__bf16)b.x; v[base + 5] = (__bf16)b.y;
    v[base + 6] = (__bf16)b.z; v[base + 7] = (__bf16)b.w;
}

// ---------------------------------------------------------------------------
// Kernel 1: one workgroup per tree.
//   GEMM D[256x128] = bf16(emb[tok]) x bf16(Wc^T) + bc   (WMMA, K-loop 4x32)
//   Wave w owns m-tiles {2w, 2w+1} x all 8 n-tiles: A fragments (the emb
//   gathers, the dominant traffic) are loaded/converted ONCE per m-tile and
//   reused across n-tiles; B (Wc) is reloaded per tile but is L2/WGP$-hot.
//   -> LDS f32 subtree buffer -> bottom-up level sums -> max over nodes,
//   clamped at 0 -> tree_enc[tree][128]
// ---------------------------------------------------------------------------
__global__ __launch_bounds__(256)
void tree_encode_kernel(const int* __restrict__ tokens,
                        const float* __restrict__ emb,
                        const float* __restrict__ Wc,
                        const float* __restrict__ bc,
                        float* __restrict__ tree_enc)
{
    __shared__ float s_sub[P_NODES * C_DIM];   // 128 KB, f32 subtree sums
    __shared__ int   s_tok[P_NODES];

    const int tid  = threadIdx.x;
    const int tree = blockIdx.x;
    const int lane = tid & 31;
    const int wave = tid >> 5;
    const int hi   = (lane >> 4) & 1;   // lane group (0: lanes 0-15, 1: 16-31)
    const int l15  = lane & 15;

    s_tok[tid] = tokens[tree * P_NODES + tid];
    __syncthreads();

    // Warm caches for this wave's two A-row groups (global_prefetch_b8)
    __builtin_prefetch(emb + (long)s_tok[(wave * 2 + 0) * 16 + l15] * E_DIM + hi * 8, 0, 3);
    __builtin_prefetch(emb + (long)s_tok[(wave * 2 + 1) * 16 + l15] * E_DIM + hi * 8, 0, 3);

    // ---- Phase A: WMMA GEMM
    #pragma unroll
    for (int mi = 0; mi < 2; ++mi) {
        const int mt   = wave * 2 + mi;
        const int mrow = mt * 16 + l15;           // A row (node index in tree)
        const int tok  = s_tok[mrow];
        const float* arow = emb + (long)tok * E_DIM;

        // Load + convert the full K=128 A strip for this m-tile once.
        v16bf af[4];
        #pragma unroll
        for (int kt = 0; kt < 4; ++kt) {
            const int kb = kt * 32;
            const float4* p0 = (const float4*)(arow + kb + hi * 16);
            const float4* p1 = (const float4*)(arow + kb + hi * 16 + 8);
            pack8_bf16(af[kt], 0, p0[0], p0[1]);
            pack8_bf16(af[kt], 8, p1[0], p1[1]);
        }

        #pragma unroll
        for (int nt = 0; nt < 8; ++nt) {
            const int ncol = nt * 16 + l15;               // B column (channel)
            const float* brow = Wc + (long)ncol * E_DIM;  // B[k][n] = Wc[n][k]

            v8f acc = {};
            #pragma unroll
            for (int kt = 0; kt < 4; ++kt) {
                const int kb = kt * 32;
                v16bf bf;
                const float4* p = (const float4*)(brow + kb + hi * 16);
                pack8_bf16(bf, 0, p[0], p[1]);
                pack8_bf16(bf, 8, p[2], p[3]);
                acc = __builtin_amdgcn_wmma_f32_16x16x32_bf16(
                    false, af[kt], false, bf, (short)0, acc, false, false);
            }

            const float bias  = bc[ncol];
            const int   mbase = mt * 16 + hi * 8;   // D layout: M = vgpr + 8*hi
            #pragma unroll
            for (int j = 0; j < 8; ++j)
                s_sub[(mbase + j) * C_DIM + ncol] = acc[j] + bias;
        }
    }
    __syncthreads();

    // ---- Phase B: bottom-up subtree sums (arity-4, parent(i) = (i-1)/4)
    // depth-3 parents: nodes 21..84 gather depth-4 children 85..255
    for (int it = tid; it < 64 * C_DIM; it += 256) {
        const int p = 21 + (it >> 7);
        const int c = it & 127;
        float s = s_sub[p * C_DIM + c];
        const int ch0 = 4 * p + 1;
        #pragma unroll
        for (int q = 0; q < 4; ++q) {
            const int ch = ch0 + q;
            if (ch < P_NODES) s += s_sub[ch * C_DIM + c];
        }
        s_sub[p * C_DIM + c] = s;
    }
    __syncthreads();
    // depth-2 parents: nodes 5..20
    for (int it = tid; it < 16 * C_DIM; it += 256) {
        const int p = 5 + (it >> 7);
        const int c = it & 127;
        float s = s_sub[p * C_DIM + c];
        const int ch0 = 4 * p + 1;
        #pragma unroll
        for (int q = 0; q < 4; ++q) s += s_sub[(ch0 + q) * C_DIM + c];
        s_sub[p * C_DIM + c] = s;
    }
    __syncthreads();
    // depth-1 parents: nodes 1..4
    for (int it = tid; it < 4 * C_DIM; it += 256) {
        const int p = 1 + (it >> 7);
        const int c = it & 127;
        float s = s_sub[p * C_DIM + c];
        const int ch0 = 4 * p + 1;
        #pragma unroll
        for (int q = 0; q < 4; ++q) s += s_sub[(ch0 + q) * C_DIM + c];
        s_sub[p * C_DIM + c] = s;
    }
    __syncthreads();
    // root
    if (tid < C_DIM) {
        float s = s_sub[tid];
        #pragma unroll
        for (int q = 1; q <= 4; ++q) s += s_sub[q * C_DIM + tid];
        s_sub[tid] = s;
    }
    __syncthreads();

    // ---- Phase C: max over nodes, floored at 0
    if (tid < C_DIM) {
        float m = 0.0f;
        for (int i = 0; i < P_NODES; ++i)
            m = fmaxf(m, s_sub[i * C_DIM + tid]);
        tree_enc[(long)tree * C_DIM + tid] = m;
    }
}

// ---------------------------------------------------------------------------
// Kernel 2: gx[dir] = tree_enc @ w_ih[dir]^T + b_ih[dir]   (WMMA bf16)
//   M=2048, N=384, K=128.  One 16x16 tile per wave.
// ---------------------------------------------------------------------------
__global__ __launch_bounds__(256)
void gru_gates_x_kernel(const float* __restrict__ tree_enc,
                        const float* __restrict__ w_ih_f, const float* __restrict__ b_ih_f,
                        const float* __restrict__ w_ih_b, const float* __restrict__ b_ih_b,
                        float* __restrict__ gx_f, float* __restrict__ gx_b)
{
    const int tid  = threadIdx.x;
    const int lane = tid & 31;
    const int wave = tid >> 5;
    const int hi   = (lane >> 4) & 1;
    const int l15  = lane & 15;

    const int MT = T_TREES / 16;   // 128
    const int NT = G3 / 16;        // 24
    const int tileId = blockIdx.x * 8 + wave;       // 0 .. 6143
    const int dir = tileId / (MT * NT);
    const int rem = tileId % (MT * NT);
    const int mt  = rem / NT;
    const int nt  = rem % NT;

    const float* w_ih = dir ? w_ih_b : w_ih_f;
    const float* b_ih = dir ? b_ih_b : b_ih_f;
    float*       gx   = dir ? gx_b   : gx_f;

    const int mrow = mt * 16 + l15;
    const int ncol = nt * 16 + l15;
    const float* arow = tree_enc + (long)mrow * C_DIM;
    const float* brow = w_ih     + (long)ncol * C_DIM;   // B[k][n] = w_ih[n][k]

    v8f acc = {};
    #pragma unroll
    for (int kt = 0; kt < 4; ++kt) {
        const int kb = kt * 32;
        v16bf af, bf;
        {
            const float4* p0 = (const float4*)(arow + kb + hi * 16);
            const float4* p1 = (const float4*)(arow + kb + hi * 16 + 8);
            pack8_bf16(af, 0, p0[0], p0[1]);
            pack8_bf16(af, 8, p1[0], p1[1]);
        }
        {
            const float4* p = (const float4*)(brow + kb + hi * 16);
            pack8_bf16(bf, 0, p[0], p[1]);
            pack8_bf16(bf, 8, p[2], p[3]);
        }
        acc = __builtin_amdgcn_wmma_f32_16x16x32_bf16(
            false, af, false, bf, (short)0, acc, false, false);
    }

    const float bias  = b_ih[ncol];
    const int   mbase = mt * 16 + hi * 8;
    #pragma unroll
    for (int j = 0; j < 8; ++j)
        gx[(long)(mbase + j) * G3 + ncol] = acc[j] + bias;
}

// ---------------------------------------------------------------------------
// Kernel 3: sequential bidirectional GRU scan + running max (single block).
//   Threads 0..383  : forward gate rows ; threads 384..767 : backward rows.
//   h vectors, gh staging and running max live in LDS; gru_out never stored.
// ---------------------------------------------------------------------------
__global__ __launch_bounds__(768)
void gru_seq_kernel(const float* __restrict__ gx_f, const float* __restrict__ gx_b,
                    const float* __restrict__ w_hh_f, const float* __restrict__ b_hh_f,
                    const float* __restrict__ w_hh_b, const float* __restrict__ b_hh_b,
                    float* __restrict__ out)
{
    __shared__ float h_f[H_DIM], h_b[H_DIM];
    __shared__ float gh[2 * G3];
    __shared__ float gmax[2 * H_DIM];

    const int tid = threadIdx.x;
    if (tid < H_DIM) { h_f[tid] = 0.0f; h_b[tid] = 0.0f; }
    if (tid < 2 * H_DIM) gmax[tid] = -3.4e38f;
    __syncthreads();

    const int dir = (tid >= G3) ? 1 : 0;
    const int r   = dir ? (tid - G3) : tid;              // gate row 0..383
    const float* wrow = (dir ? w_hh_b : w_hh_f) + (long)r * H_DIM;
    const float  bias = (dir ? b_hh_b : b_hh_f)[r];
    const float* hvec = dir ? h_b : h_f;

    const bool doF = (tid < H_DIM);
    const bool doB = (tid >= G3) && (tid < G3 + H_DIM);

    for (int t = 0; t < T_TREES; ++t) {
        // gh = w_hh @ h + b_hh  (one row per thread, h broadcast from LDS)
        float acc = bias;
        #pragma unroll 8
        for (int k = 0; k < H_DIM; k += 4) {
            const float4 w4 = *(const float4*)(wrow + k);
            acc += w4.x * hvec[k]     + w4.y * hvec[k + 1]
                 + w4.z * hvec[k + 2] + w4.w * hvec[k + 3];
        }
        gh[tid] = acc;
        __syncthreads();

        if (doF | doB) {
            const int c = doF ? tid : (tid - G3);
            const float* gxrow = doF ? (gx_f + (long)t * G3)
                                     : (gx_b + (long)(T_TREES - 1 - t) * G3);
            float*       hcur  = doF ? h_f : h_b;
            const float* ghp   = gh + (doF ? 0 : G3);
            const float rg = sigmoidf_(gxrow[c]             + ghp[c]);
            const float zg = sigmoidf_(gxrow[H_DIM + c]     + ghp[H_DIM + c]);
            const float ng = tanhf    (gxrow[2 * H_DIM + c] + rg * ghp[2 * H_DIM + c]);
            const float h2 = (1.0f - zg) * ng + zg * hcur[c];
            hcur[c] = h2;
            const int gi = (doF ? 0 : H_DIM) + c;
            gmax[gi] = fmaxf(gmax[gi], h2);
        }
        __syncthreads();
    }

    if (tid < 2 * H_DIM) out[tid] = gmax[tid];
}

// ---------------------------------------------------------------------------
extern "C" void kernel_launch(void* const* d_in, const int* in_sizes, int n_in,
                              void* d_out, int out_size, void* d_ws, size_t ws_size,
                              hipStream_t stream)
{
    const int*   tokens = (const int*)  d_in[0];
    // d_in[1]=parent, d_in[2]=depth, d_in[3]=tree_id: structure is fixed, unused
    const float* emb    = (const float*)d_in[4];
    const float* Wc     = (const float*)d_in[5];
    const float* bc     = (const float*)d_in[6];
    const float* w_ih_f = (const float*)d_in[7];
    const float* w_hh_f = (const float*)d_in[8];
    const float* b_ih_f = (const float*)d_in[9];
    const float* b_hh_f = (const float*)d_in[10];
    const float* w_ih_b = (const float*)d_in[11];
    const float* w_hh_b = (const float*)d_in[12];
    const float* b_ih_b = (const float*)d_in[13];
    const float* b_hh_b = (const float*)d_in[14];

    float* tree_enc = (float*)d_ws;                       // 2048*128 f32 (1 MB)
    float* gx_f     = tree_enc + (size_t)T_TREES * C_DIM; // 2048*384 f32 (3 MB)
    float* gx_b     = gx_f     + (size_t)T_TREES * G3;    // 2048*384 f32 (3 MB)

    hipLaunchKernelGGL(tree_encode_kernel, dim3(T_TREES), dim3(256), 0, stream,
                       tokens, emb, Wc, bc, tree_enc);

    hipLaunchKernelGGL(gru_gates_x_kernel, dim3(768), dim3(256), 0, stream,
                       tree_enc, w_ih_f, b_ih_f, w_ih_b, b_ih_b, gx_f, gx_b);

    hipLaunchKernelGGL(gru_seq_kernel, dim3(1), dim3(2 * G3), 0, stream,
                       gx_f, gx_b, w_hh_f, b_hh_f, w_hh_b, b_hh_b, (float*)d_out);
}
